// Region_Proposal_Network_76940044140545
// MI455X (gfx1250) — matrix operand order repaired and checked by reading, not compiled, not run on hardware
//
#include <hip/hip_runtime.h>
#include <hip/hip_bf16.h>

// ---------------- problem constants ----------------
#define HH     50
#define WW2    84
#define NPIX   4200          // HH*WW2
#define MPAD   4224          // NPIX padded to multiple of 32
#define MT1    132           // conv M tiles (32 rows each)
#define MT2    264           // gemm2 M tiles (16 rows each)
#define CIN    512
#define CMID   512
#define KTAP   4608          // 9*512
#define NIMG   2
#define NANCH  9
#define KA     37800         // NPIX*NANCH
#define NSORT  65536
#define NPRE   6000
#define NPOST  300
#define NEGV   (-1e9f)

// ---------------- d_out layout (float elements) ----------------
#define OFF_LOC    0                    // 2*37800*4 = 302400
#define OFF_SCORE  302400               // 2*37800*2 = 151200
#define OFF_ROIS   453600               // 2*300*4   = 2400
#define OFF_RIDX   456000               // 2*300     = 600 (int32 bits)
#define OFF_ANCH   456600               // 37800*4   = 151200
// total = 607800

// ---------------- workspace layout (bytes) ----------------
#define XT_OFF   ((size_t)0)                       // f16 x, NHWC: 2*4200*512*2 = 8,601,600
#define WB1_OFF  ((size_t)8601600)                 // f16 conv1 w [co][tap*512+ci]: 4,718,592
#define HID_OFF  ((size_t)13320192)                // f16 hidden [img][m(4224)][c]: 8,650,752
#define WB2_OFF  ((size_t)21970944)                // f16 packed 1x1 w [n(64)][k(512)]: 65,536
#define ZPG_OFF  ((size_t)22036480)                // f16 zero page: 1,024
#define BOX_OFF  ((size_t)22037504)                // f32 boxes [2][37800][4]: 1,209,600
#define MSC_OFF  ((size_t)23247104)                // f32 masked scores [2][37800]: 302,400
#define ARR_OFF  ((size_t)23549504)                // u64 sort keys [2][65536]: 1,048,576
// total = 24,598,080

typedef __attribute__((ext_vector_type(16))) _Float16 v16h;
typedef __attribute__((ext_vector_type(8)))  _Float16 v8h;
typedef __attribute__((ext_vector_type(8)))  float    v8f;

__device__ __forceinline__ v16h ld_a16(const _Float16* p) {
  // A per-lane layout: elems 0..15 = K k0+hi*16..+15
  v8h lo = *(const v8h*)(p);
  v8h hi = *(const v8h*)(p + 8);
  return __builtin_shufflevector(lo, hi, 0,1,2,3,4,5,6,7,8,9,10,11,12,13,14,15);
}

__device__ __forceinline__ v8f wmma_f16(v16h a, v16h b, v8f c) {
  return __builtin_amdgcn_wmma_f32_16x16x32_f16(false, a, false, b, (short)0, c, false, false);
}

// ---------------- packing kernels ----------------
// x [2][512][50][84] f32  ->  xt [2][4200][512] f16 (NHWC)
__global__ void rpn_pack_x(const float* __restrict__ x, _Float16* __restrict__ xt) {
  size_t t = (size_t)blockIdx.x * blockDim.x + threadIdx.x;
  if (t >= (size_t)NIMG * NPIX * CIN) return;
  int img = (int)(t / ((size_t)NPIX * CIN));
  int rem = (int)(t - (size_t)img * NPIX * CIN);
  int pix = rem / CIN, ci = rem % CIN;
  xt[t] = (_Float16)x[(size_t)img * CIN * NPIX + (size_t)ci * NPIX + pix];
}

// conv1_w [512][512][3][3] -> wb1 [co][tap*512+ci] f16 (K = tap-major, ci-minor)
__global__ void rpn_pack_w1(const float* __restrict__ w, _Float16* __restrict__ wb1) {
  size_t t = (size_t)blockIdx.x * blockDim.x + threadIdx.x;
  if (t >= (size_t)CMID * KTAP) return;
  int co = (int)(t / KTAP);
  int rem = (int)(t - (size_t)co * KTAP);
  int tap = rem / CIN, ci = rem % CIN;
  wb1[t] = (_Float16)w[((size_t)co * CIN + ci) * 9 + tap];
}

// loc_w [36][512], score_w [18][512] -> wb2 [n(64)][k(512)] f16 (n>=54 zero)
__global__ void rpn_pack_w2(const float* __restrict__ loc_w, const float* __restrict__ score_w,
                            _Float16* __restrict__ wb2) {
  int t = blockIdx.x * blockDim.x + threadIdx.x;
  if (t >= 64 * CMID) return;
  int n = t / CMID, k = t % CMID;
  float v = 0.f;
  if (n < 36)      v = loc_w[(size_t)n * CMID + k];
  else if (n < 54) v = score_w[(size_t)(n - 36) * CMID + k];
  wb2[t] = (_Float16)v;
}

__global__ void rpn_zero_page(float* __restrict__ zp) {
  zp[threadIdx.x] = 0.f;   // 256 floats = 1024 B
}

// ---------------- conv1 3x3 as implicit-GEMM WMMA ----------------
// Wave tile: 32M x 64N (2 A-frags x 4 B-frags -> 8 WMMA per K-step).
// grid (132, 2, 2), block (32,4): wid = by*4+ty in 0..7 covers cols wid*64..+63.
// OOB A rows read a zero page -> branchless inner loop, immediate-offset loads.
__global__ void __launch_bounds__(128)
rpn_conv1_wmma(const _Float16* __restrict__ xt, const _Float16* __restrict__ wb1,
               const float* __restrict__ bias, const _Float16* __restrict__ zpage,
               _Float16* __restrict__ hidden) {
  const int mtile = blockIdx.x;                       // 0..131
  const int wid   = blockIdx.y * 4 + threadIdx.y;     // 0..7
  const int img   = blockIdx.z;
  const int lane  = threadIdx.x;
  const int r  = lane & 15;
  const int hi = lane >> 4;
  const int m0 = mtile * 32;
  const int n0 = wid * 64;

  v8f z8 = {0.f,0.f,0.f,0.f,0.f,0.f,0.f,0.f};
  v8f acc[2][4];
#pragma unroll
  for (int s = 0; s < 2; ++s)
#pragma unroll
    for (int t = 0; t < 4; ++t) acc[s][t] = z8;

  const int mm0 = m0 + r, mm1 = m0 + 16 + r;
  const bool ok0 = mm0 < NPIX, ok1 = mm1 < NPIX;
  const int ph0 = ok0 ? mm0 / WW2 : 0, pw0 = ok0 ? mm0 % WW2 : 0;
  const int ph1 = ok1 ? mm1 / WW2 : 0, pw1 = ok1 ? mm1 % WW2 : 0;

  const _Float16* xim = xt + (size_t)img * NPIX * CIN + hi * 16;
  const _Float16* zp  = zpage + hi * 16;
  const _Float16* bb0 = wb1 + (size_t)(n0 +  0 + r) * KTAP + hi * 16;
  const _Float16* bb1 = wb1 + (size_t)(n0 + 16 + r) * KTAP + hi * 16;
  const _Float16* bb2 = wb1 + (size_t)(n0 + 32 + r) * KTAP + hi * 16;
  const _Float16* bb3 = wb1 + (size_t)(n0 + 48 + r) * KTAP + hi * 16;

  for (int tap = 0; tap < 9; ++tap) {
    const int dh = tap / 3 - 1, dw = tap % 3 - 1;
    const int sh0 = ph0 + dh, sw0 = pw0 + dw;
    const int sh1 = ph1 + dh, sw1 = pw1 + dw;
    const bool v0 = ok0 && (unsigned)sh0 < (unsigned)HH && (unsigned)sw0 < (unsigned)WW2;
    const bool v1 = ok1 && (unsigned)sh1 < (unsigned)HH && (unsigned)sw1 < (unsigned)WW2;
    const _Float16* a0 = v0 ? (xim + (size_t)(sh0 * WW2 + sw0) * CIN) : zp;
    const _Float16* a1 = v1 ? (xim + (size_t)(sh1 * WW2 + sw1) * CIN) : zp;
    const _Float16* b0 = bb0 + tap * CIN;
    const _Float16* b1 = bb1 + tap * CIN;
    const _Float16* b2 = bb2 + tap * CIN;
    const _Float16* b3 = bb3 + tap * CIN;
    __builtin_prefetch(b0 + CIN, 0, 1);
    __builtin_prefetch(b1 + CIN, 0, 1);
    __builtin_prefetch(b2 + CIN, 0, 1);
    __builtin_prefetch(b3 + CIN, 0, 1);
#pragma unroll 4
    for (int kc = 0; kc < 16; ++kc) {
      const int o = kc * 32;
      v16h av0 = ld_a16(a0 + o);
      v16h av1 = ld_a16(a1 + o);
      v16h bv0 = *(const v16h*)(b0 + o);
      v16h bv1 = *(const v16h*)(b1 + o);
      v16h bv2 = *(const v16h*)(b2 + o);
      v16h bv3 = *(const v16h*)(b3 + o);
      acc[0][0] = wmma_f16(av0, bv0, acc[0][0]);
      acc[0][1] = wmma_f16(av0, bv1, acc[0][1]);
      acc[0][2] = wmma_f16(av0, bv2, acc[0][2]);
      acc[0][3] = wmma_f16(av0, bv3, acc[0][3]);
      acc[1][0] = wmma_f16(av1, bv0, acc[1][0]);
      acc[1][1] = wmma_f16(av1, bv1, acc[1][1]);
      acc[1][2] = wmma_f16(av1, bv2, acc[1][2]);
      acc[1][3] = wmma_f16(av1, bv3, acc[1][3]);
    }
  }

  const float bs0 = bias[n0 +  0 + r];
  const float bs1 = bias[n0 + 16 + r];
  const float bs2 = bias[n0 + 32 + r];
  const float bs3 = bias[n0 + 48 + r];
  const float bs[4] = {bs0, bs1, bs2, bs3};
  _Float16* hbase = hidden + (size_t)img * MPAD * CMID;
#pragma unroll
  for (int s = 0; s < 2; ++s) {
#pragma unroll
    for (int t = 0; t < 4; ++t) {
#pragma unroll
      for (int v = 0; v < 8; ++v) {
        int mm = m0 + s * 16 + v + hi * 8;            // D: M = v + 8*hi, N = lane&15
        float val = fmaxf(acc[s][t][v] + bs[t], 0.0f); // bias + ReLU
        hbase[(size_t)mm * CMID + (n0 + 16 * t + r)] = (_Float16)val;
      }
    }
  }
}

// ---------------- fused 1x1 loc+score GEMM (K=512, N=64) ----------------
// grid (264, 2), block (32,4): ntile = threadIdx.y covers 64 output channels
__global__ void __launch_bounds__(128)
rpn_gemm2_wmma(const _Float16* __restrict__ hidden, const _Float16* __restrict__ wb2,
               const float* __restrict__ loc_b, const float* __restrict__ score_b,
               float* __restrict__ out) {
  const int mtile = blockIdx.x;
  const int img   = blockIdx.y;
  const int ntile = threadIdx.y;
  const int lane  = threadIdx.x;
  const int r = lane & 15, hi = lane >> 4;
  const int m0 = mtile * 16, n0 = ntile * 16;

  v8f acc = {0.f,0.f,0.f,0.f,0.f,0.f,0.f,0.f};
  const _Float16* arow = hidden + ((size_t)img * MPAD + m0 + r) * CMID + hi * 16;
  const _Float16* brow = wb2 + (size_t)(n0 + r) * CMID + hi * 16;
#pragma unroll
  for (int kt = 0; kt < 16; ++kt) {
    v16h av = ld_a16(arow + kt * 32);
    v16h bv = *(const v16h*)(brow + kt * 32);
    acc = wmma_f16(av, bv, acc);
  }
  const int col = n0 + r;
  float bs = (col < 36) ? loc_b[col] : ((col < 54) ? score_b[col - 36] : 0.f);
#pragma unroll
  for (int v = 0; v < 8; ++v) {
    int mm = m0 + v + hi * 8;
    if (mm < NPIX) {
      float val = acc[v] + bs;
      if (col < 36)
        out[OFF_LOC + ((size_t)img * NPIX + mm) * 36 + col] = val;          // rpn_locs
      else if (col < 54)
        out[OFF_SCORE + ((size_t)img * NPIX + mm) * 18 + (col - 36)] = val; // rpn_scores (raw)
    }
  }
}

// ---------------- anchors ----------------
__global__ void rpn_anchors(float* __restrict__ out) {
  int i = blockIdx.x * blockDim.x + threadIdx.x;
  if (i >= KA) return;
  int m = i / NANCH, a = i % NANCH;
  int ri = a / 3, si = a % 3;
  double ratio = (ri == 0) ? 0.5 : ((ri == 1) ? 1.0 : 2.0);
  double scale = (si == 0) ? 8.0 : ((si == 1) ? 16.0 : 32.0);
  double h = 16.0 * scale * sqrt(ratio);
  double w = 16.0 * scale * sqrt(1.0 / ratio);
  float y1 = (float)(8.0 - h * 0.5), x1 = (float)(8.0 - w * 0.5);
  float y2 = (float)(8.0 + h * 0.5), x2 = (float)(8.0 + w * 0.5);
  double sy = (double)((m / WW2) * 16);
  double sx = (double)((m % WW2) * 16);
  float* o = out + OFF_ANCH + (size_t)i * 4;
  o[0] = (float)(sy + (double)y1);
  o[1] = (float)(sx + (double)x1);
  o[2] = (float)(sy + (double)y2);
  o[3] = (float)(sx + (double)x2);
}

// ---------------- roi_indices (int32 bits into float buffer) ----------------
__global__ void rpn_roi_idx(int* __restrict__ outi) {
  int t = blockIdx.x * blockDim.x + threadIdx.x;
  if (t >= NIMG * NPOST) return;
  outi[OFF_RIDX + t] = t / NPOST;
}

// ---------------- decode boxes, softmax fg, size-mask, sort keys ----------------
__global__ void rpn_prep(const float* __restrict__ out, const int* __restrict__ img_h,
                         const int* __restrict__ img_w, float* __restrict__ boxes,
                         float* __restrict__ mscore, unsigned long long* __restrict__ arr) {
  int t = blockIdx.x * blockDim.x + threadIdx.x;
  if (t >= NIMG * NSORT) return;
  int img = t / NSORT;
  int i = t - img * NSORT;
  if (i >= KA) { arr[(size_t)img * NSORT + i] = ~0ULL; return; }   // padding sorts last
  size_t gi = (size_t)img * KA + i;
  const float* loc = out + OFF_LOC + gi * 4;
  const float* anc = out + OFF_ANCH + (size_t)i * 4;
  float h  = anc[2] - anc[0], w = anc[3] - anc[1];
  float yc = anc[0] + 0.5f * h, xc = anc[1] + 0.5f * w;
  float cy = loc[0] * h + yc, cx = loc[1] * w + xc;
  float nh = expf(loc[2]) * h, nw = expf(loc[3]) * w;
  float H = (float)img_h[0], W = (float)img_w[0];
  float y1 = fminf(fmaxf(cy - 0.5f * nh, 0.f), H);
  float x1 = fminf(fmaxf(cx - 0.5f * nw, 0.f), W);
  float y2 = fminf(fmaxf(cy + 0.5f * nh, 0.f), H);
  float x2 = fminf(fmaxf(cx + 0.5f * nw, 0.f), W);
  bool valid = ((y2 - y1) >= 16.f) && ((x2 - x1) >= 16.f);
  const float* sc = out + OFF_SCORE + gi * 2;
  float s0 = sc[0], s1 = sc[1];
  float mx = fmaxf(s0, s1);
  float e0 = expf(s0 - mx), e1 = expf(s1 - mx);
  float score = valid ? (e1 / (e0 + e1)) : NEGV;
  float* b = boxes + gi * 4;
  b[0] = y1; b[1] = x1; b[2] = y2; b[3] = x2;
  mscore[gi] = score;
  unsigned u = __float_as_uint(score);
  unsigned key = (u & 0x80000000u) ? u : ~(u | 0x80000000u);   // ascending key == descending score
  arr[(size_t)img * NSORT + i] = ((unsigned long long)key << 32) | (unsigned)i;
}

// ---------------- global bitonic sort step (ascending) ----------------
__global__ void rpn_bitonic(unsigned long long* __restrict__ arr, unsigned k, unsigned j) {
  unsigned i = blockIdx.x * blockDim.x + threadIdx.x;   // 0..65535
  unsigned long long* a = arr + (size_t)blockIdx.y * NSORT;
  unsigned ixj = i ^ j;
  if (ixj > i) {
    unsigned long long x = a[i], y = a[ixj];
    bool up = ((i & k) == 0);
    if ((x > y) == up) { a[i] = y; a[ixj] = x; }
  }
}

// ---------------- greedy NMS (one block per image) + collect 300 ----------------
__global__ void rpn_nms(const unsigned long long* __restrict__ arr,
                        const float* __restrict__ boxes, const float* __restrict__ mscore,
                        float* __restrict__ out) {
  const int img = blockIdx.x;
  extern __shared__ unsigned char smem[];
  float4* bb = (float4*)smem;                                   // NPRE * 16 bytes
  unsigned char* supp = smem + (size_t)NPRE * sizeof(float4);   // NPRE bytes
  const unsigned long long* a = arr + (size_t)img * NSORT;

  for (int i = threadIdx.x; i < NPRE; i += blockDim.x) {
    unsigned idx = (unsigned)(a[i] & 0xFFFFFFFFu);
    const float* bp = boxes + ((size_t)img * KA + idx) * 4;
    bb[i] = make_float4(bp[0], bp[1], bp[2], bp[3]);
    supp[i] = (mscore[(size_t)img * KA + idx] <= NEGV * 0.5f) ? 1 : 0;
  }
  __syncthreads();

  for (int i = 0; i < NPRE; ++i) {
    if (!supp[i]) {                                   // uniform branch (shared read)
      float4 bi = bb[i];
      float ai = (bi.z - bi.x) * (bi.w - bi.y);
      for (int j = i + 1 + threadIdx.x; j < NPRE; j += blockDim.x) {
        float4 bj = bb[j];
        float aj = (bj.z - bj.x) * (bj.w - bj.y);
        float ty = fmaxf(bi.x, bj.x), tx = fmaxf(bi.y, bj.y);
        float by = fminf(bi.z, bj.z), bx = fminf(bi.w, bj.w);
        float ih = fmaxf(by - ty, 0.f), iw = fmaxf(bx - tx, 0.f);
        float inter = ih * iw;
        float iou = inter / (ai + aj - inter + 1e-9f);
        if (iou > 0.7f) supp[j] = 1;
      }
    }
    __syncthreads();
  }

  if (threadIdx.x == 0) {
    int nk = 0;
    for (int i = 0; i < NPRE; ++i) {
      if (!supp[i]) {
        if (nk < NPOST) {
          float4 b = bb[i];
          float* o = out + OFF_ROIS + ((size_t)img * NPOST + nk) * 4;
          o[0] = b.x; o[1] = b.y; o[2] = b.z; o[3] = b.w;
        }
        ++nk;
      }
    }
    for (int r2 = nk; r2 < NPOST; ++r2) {
      float* o = out + OFF_ROIS + ((size_t)img * NPOST + r2) * 4;
      o[0] = 0.f; o[1] = 0.f; o[2] = 0.f; o[3] = 0.f;
    }
  }
}

// ---------------- launcher ----------------
extern "C" void kernel_launch(void* const* d_in, const int* in_sizes, int n_in,
                              void* d_out, int out_size, void* d_ws, size_t ws_size,
                              hipStream_t stream) {
  (void)in_sizes; (void)n_in; (void)out_size; (void)ws_size;
  const float* x       = (const float*)d_in[0];
  const float* conv1_w = (const float*)d_in[1];
  const float* conv1_b = (const float*)d_in[2];
  const float* score_w = (const float*)d_in[3];
  const float* score_b = (const float*)d_in[4];
  const float* loc_w   = (const float*)d_in[5];
  const float* loc_b   = (const float*)d_in[6];
  const int*   img_h   = (const int*)d_in[7];
  const int*   img_w   = (const int*)d_in[8];
  float* out = (float*)d_out;
  char* ws = (char*)d_ws;

  _Float16* xt     = (_Float16*)(ws + XT_OFF);
  _Float16* wb1    = (_Float16*)(ws + WB1_OFF);
  _Float16* hidden = (_Float16*)(ws + HID_OFF);
  _Float16* wb2    = (_Float16*)(ws + WB2_OFF);
  _Float16* zpage  = (_Float16*)(ws + ZPG_OFF);
  float*    boxes  = (float*)(ws + BOX_OFF);
  float*    mscore = (float*)(ws + MSC_OFF);
  unsigned long long* arr = (unsigned long long*)(ws + ARR_OFF);

  // pack inputs to f16 GEMM layouts + independent small outputs
  rpn_pack_x<<<(NIMG * NPIX * CIN + 255) / 256, 256, 0, stream>>>(x, xt);
  rpn_pack_w1<<<(CMID * KTAP + 255) / 256, 256, 0, stream>>>(conv1_w, wb1);
  rpn_pack_w2<<<(64 * CMID + 255) / 256, 256, 0, stream>>>(loc_w, score_w, wb2);
  rpn_zero_page<<<1, 256, 0, stream>>>((float*)zpage);
  rpn_anchors<<<(KA + 255) / 256, 256, 0, stream>>>(out);
  rpn_roi_idx<<<(NIMG * NPOST + 255) / 256, 256, 0, stream>>>((int*)d_out);

  // conv1 (implicit GEMM, WMMA, 32x64 wave tiles) -> hidden (f16, relu+bias fused)
  rpn_conv1_wmma<<<dim3(MT1, 2, NIMG), dim3(32, 4, 1), 0, stream>>>(xt, wb1, conv1_b, zpage, hidden);

  // fused loc+score 1x1 conv (WMMA) -> rpn_locs / rpn_scores in d_out
  rpn_gemm2_wmma<<<dim3(MT2, NIMG, 1), dim3(32, 4, 1), 0, stream>>>(hidden, wb2, loc_b, score_b, out);

  // decode + softmax + mask + sort keys
  rpn_prep<<<(NIMG * NSORT + 255) / 256, 256, 0, stream>>>(out, img_h, img_w, boxes, mscore, arr);

  // full bitonic sort (both images via grid.y)
  for (unsigned k = 2; k <= NSORT; k <<= 1)
    for (unsigned j = k >> 1; j >= 1; j >>= 1)
      rpn_bitonic<<<dim3(NSORT / 256, NIMG, 1), 256, 0, stream>>>(arr, k, j);

  // greedy NMS + collect rois (LDS: 6000 boxes + 6000 flags = 102000 B)
  size_t shmem = (size_t)NPRE * sizeof(float4) + NPRE;
  rpn_nms<<<dim3(NIMG, 1, 1), 256, shmem, stream>>>(arr, boxes, mscore, out);
}
